// GraphToGraph_60833916780728
// MI455X (gfx1250) — matrix operand order repaired and checked by reading, not compiled, not run on hardware
//
#include <hip/hip_runtime.h>
#include <hip/hip_fp16.h>

#define N_NODES 50000
#define N_EDGES 1600000
#define D_FEAT  256
#define D_HID   256
#define D_EDGE  64
#define BM      128   // node rows per workgroup (8 waves x 16 rows)

typedef _Float16 h8   __attribute__((ext_vector_type(8)));
typedef _Float16 v16h __attribute__((ext_vector_type(16)));
typedef float    v8f  __attribute__((ext_vector_type(8)));
typedef float    f4   __attribute__((ext_vector_type(4)));

union V16 { v16h v; h8 h[2]; };

// ---------------------------------------------------------------------------
// Kernel 0: transpose + convert weights to f16, column-major (WT[n][k]) so
// each WMMA B-fragment is 2 contiguous 16B loads per lane.
// ---------------------------------------------------------------------------
__global__ __launch_bounds__(256) void prep_weights_kernel(
    const float* __restrict__ Wn1, const float* __restrict__ We1,
    const float* __restrict__ We2,
    _Float16* __restrict__ WT1, _Float16* __restrict__ WT2) {
  int n = blockIdx.x, k = threadIdx.x;
  if (n < 256)      WT1[n * 256 + k] = (_Float16)Wn1[k * 256 + n];
  else if (n < 512) WT1[n * 256 + k] = (_Float16)We1[k * 256 + (n - 256)];
  else              WT2[(n - 512) * 256 + k] = (_Float16)We2[k * 64 + (n - 512)];
}

// ---- helpers --------------------------------------------------------------
__device__ __forceinline__ void load_btile(const _Float16* __restrict__ wcol,
                                           V16 b[8]) {
#pragma unroll
  for (int ks = 0; ks < 8; ++ks) {
    b[ks].h[0] = *(const h8*)(wcol + ks * 32);
    b[ks].h[1] = *(const h8*)(wcol + ks * 32 + 16);
  }
}

__device__ __forceinline__ v8f mma_tile(const v16h a[8], const V16 b[8]) {
  v8f acc = {};
#pragma unroll
  for (int ks = 0; ks < 8; ++ks)
    acc = __builtin_amdgcn_wmma_f32_16x16x32_f16(false, a[ks], false, b[ks].v,
                                                 (short)0, acc, false, false);
  return acc;
}

// ---------------------------------------------------------------------------
// Kernel 1: fused node pipeline. Reads X once; WMMA for both 256-wide GEMMs
// and the 64-wide edge-feature GEMM. B tiles are double-buffered in registers
// with a sched_barrier so next-tile loads stay in flight across the current
// tile's WMMAs (1-deep software pipeline).
// ---------------------------------------------------------------------------
__global__ __launch_bounds__(256) void fused_node_kernel(
    const float* __restrict__ X,
    const _Float16* __restrict__ WT1,
    const _Float16* __restrict__ WT2,
    const float* __restrict__ bn1, const float* __restrict__ Wn2,
    const float* __restrict__ bn2, const float* __restrict__ be1,
    const float* __restrict__ be2,
    float* __restrict__ nodeOut, _Float16* __restrict__ ef) {
  // per-wave 16x256 f16 hidden buffer, k-blocked: idx = (k>>3)*128 + m*8 + (k&7)
  __shared__ __align__(16) _Float16 ldsH2[8 * 16 * 256];  // 64 KB

  const int tid  = threadIdx.x;
  const int w    = tid >> 5;     // wave id
  const int lane = tid & 31;
  const int l16  = lane & 15;
  const int hi   = lane >> 4;    // lane-half selector

  const int rowBase = blockIdx.x * BM + w * 16;

  // ---- build + cache 8 A-fragments (K=0..255) from global f32 -> f16 ----
  int rowG = rowBase + l16;
  int rowC = rowG < N_NODES ? rowG : (N_NODES - 1);
  const float* rp = X + (size_t)rowC * D_FEAT;
  v16h aF[8];
#pragma unroll
  for (int ks = 0; ks < 8; ++ks) {
    int k0 = ks * 32 + hi * 8;           // A: lanes16-31 start at K=+8
    f4 x0 = *(const f4*)(rp + k0);
    f4 x1 = *(const f4*)(rp + k0 + 4);
    f4 x2 = *(const f4*)(rp + k0 + 16);
    f4 x3 = *(const f4*)(rp + k0 + 20);
    V16 u;
#pragma unroll
    for (int i = 0; i < 4; ++i) {
      u.h[0][i]     = (_Float16)x0[i];
      u.h[0][4 + i] = (_Float16)x1[i];
      u.h[1][i]     = (_Float16)x2[i];
      u.h[1][4 + i] = (_Float16)x3[i];
    }
    aF[ks] = u.v;
  }

  float partial[8];
#pragma unroll
  for (int r = 0; r < 8; ++r) partial[r] = 0.f;

  _Float16* myH2 = ldsH2 + w * (16 * 256);

  // per-lane base column pointer; tile stride = 16 cols * 256 halves
  const _Float16* wbase = WT1 + (size_t)l16 * 256 + hi * 8;

  // ---- GEMM1: 32 col tiles (0..15 node-hidden, 16..31 edge-hidden) ----
  V16 b0[8], b1[8];
  load_btile(wbase, b0);

#pragma unroll 1
  for (int ct = 0; ct < 32; ct += 2) {
    // issue loads for tile ct+1 into the other buffer, then fence so the
    // scheduler cannot sink them below the current tile's WMMAs
    load_btile(wbase + (size_t)(ct + 1) * (16 * 256), b1);
    __builtin_amdgcn_sched_barrier(0);
    {
      v8f acc = mma_tile(aF, b0);
      if (ct < 16) {
        int nn   = ct * 16 + l16;
        float bv = bn1[nn];
        float w2 = Wn2[nn];
#pragma unroll
        for (int r = 0; r < 8; ++r) {
          float v = acc[r] + bv;
          v = v > 0.f ? v : 0.f;
          partial[r] += v * w2;
        }
      } else {
        int col  = (ct - 16) * 16 + l16;
        float bv = be1[col];
#pragma unroll
        for (int r = 0; r < 8; ++r) {
          float v = acc[r] + bv;
          v = v > 0.f ? v : 0.f;
          int m = r + hi * 8;
          myH2[((col >> 3) << 7) + (m << 3) + (col & 7)] = (_Float16)v;
        }
      }
    }
    if (ct + 2 < 32) {
      load_btile(wbase + (size_t)(ct + 2) * (16 * 256), b0);
      __builtin_amdgcn_sched_barrier(0);
    }
    {
      v8f acc = mma_tile(aF, b1);
      int cto = ct + 1;
      if (cto < 16) {
        int nn   = cto * 16 + l16;
        float bv = bn1[nn];
        float w2 = Wn2[nn];
#pragma unroll
        for (int r = 0; r < 8; ++r) {
          float v = acc[r] + bv;
          v = v > 0.f ? v : 0.f;
          partial[r] += v * w2;
        }
      } else {
        int col  = (cto - 16) * 16 + l16;
        float bv = be1[col];
#pragma unroll
        for (int r = 0; r < 8; ++r) {
          float v = acc[r] + bv;
          v = v > 0.f ? v : 0.f;
          int m = r + hi * 8;
          myH2[((col >> 3) << 7) + (m << 3) + (col & 7)] = (_Float16)v;
        }
      }
    }
  }

  // ---- node scores: reduce across 16-lane groups, sigmoid, store ----
  float b2 = bn2[0];
#pragma unroll
  for (int r = 0; r < 8; ++r) {
    float p = partial[r];
    p += __shfl_xor(p, 8, 16);
    p += __shfl_xor(p, 4, 16);
    p += __shfl_xor(p, 2, 16);
    p += __shfl_xor(p, 1, 16);
    if (l16 == 0) {
      int row = rowBase + r + hi * 8;
      if (row < N_NODES) nodeOut[row] = 1.f / (1.f + __expf(-(p + b2)));
    }
  }

  __syncthreads();

  // ---- GEMM2: edge_feats = hidden @ We2 + be2 (K=256, N=64) ----
  v16h aE[8];
#pragma unroll
  for (int ks = 0; ks < 8; ++ks) {
    int k0 = ks * 32 + hi * 8;
    V16 u;
    u.h[0] = *(const h8*)(myH2 + ((k0 >> 3) << 7) + (l16 << 3));
    u.h[1] = *(const h8*)(myH2 + (((k0 + 16) >> 3) << 7) + (l16 << 3));
    aE[ks] = u.v;
  }
  const _Float16* wbase2 = WT2 + (size_t)l16 * 256 + hi * 8;
  load_btile(wbase2, b0);
#pragma unroll 1
  for (int ct = 0; ct < 4; ct += 2) {
    load_btile(wbase2 + (size_t)(ct + 1) * (16 * 256), b1);
    __builtin_amdgcn_sched_barrier(0);
    {
      v8f acc = mma_tile(aE, b0);
      int col  = ct * 16 + l16;
      float bv = be2[col];
#pragma unroll
      for (int r = 0; r < 8; ++r) {
        int row = rowBase + r + hi * 8;
        if (row < N_NODES)
          ef[(size_t)row * D_EDGE + col] = (_Float16)(acc[r] + bv);
      }
    }
    if (ct + 2 < 4) {
      load_btile(wbase2 + (size_t)(ct + 2) * (16 * 256), b0);
      __builtin_amdgcn_sched_barrier(0);
    }
    {
      v8f acc = mma_tile(aE, b1);
      int col  = (ct + 1) * 16 + l16;
      float bv = be2[col];
#pragma unroll
      for (int r = 0; r < 8; ++r) {
        int row = rowBase + r + hi * 8;
        if (row < N_NODES)
          ef[(size_t)row * D_EDGE + col] = (_Float16)(acc[r] + bv);
      }
    }
  }
}

// ---------------------------------------------------------------------------
// Kernel 2: per-edge gather + dot + sigmoid. 4 lanes per edge, f16 table is
// L2-resident (6.4 MB << 192 MB).
// ---------------------------------------------------------------------------
__global__ __launch_bounds__(256) void edge_kernel(
    const int* __restrict__ ids, const _Float16* __restrict__ ef,
    float* __restrict__ out) {
  int t = blockIdx.x * 256 + threadIdx.x;
  int e = t >> 2;
  int p = t & 3;
  if (e >= N_EDGES) return;
  int s = ids[e];
  int d = ids[N_EDGES + e];
  const h8* sp = (const h8*)(ef + (size_t)s * D_EDGE + p * 16);
  const h8* dp = (const h8*)(ef + (size_t)d * D_EDGE + p * 16);
  h8 s0 = sp[0], s1 = sp[1];
  h8 d0 = dp[0], d1 = dp[1];
  float acc = 0.f;
#pragma unroll
  for (int i = 0; i < 8; ++i)
    acc += (float)s0[i] * (float)d0[i] + (float)s1[i] * (float)d1[i];
  acc += __shfl_xor(acc, 1, 4);
  acc += __shfl_xor(acc, 2, 4);
  if (p == 0) out[N_NODES + e] = 1.f / (1.f + __expf(-acc));
}

// ---------------------------------------------------------------------------
extern "C" void kernel_launch(void* const* d_in, const int* in_sizes, int n_in,
                              void* d_out, int out_size, void* d_ws, size_t ws_size,
                              hipStream_t stream) {
  const float* X   = (const float*)d_in[0];
  // d_in[1] node_xy: unused by the reference outputs
  const int*   ids = (const int*)d_in[2];
  const float* Wn1 = (const float*)d_in[3];
  const float* bn1 = (const float*)d_in[4];
  const float* Wn2 = (const float*)d_in[5];
  const float* bn2 = (const float*)d_in[6];
  const float* We1 = (const float*)d_in[7];
  const float* be1 = (const float*)d_in[8];
  const float* We2 = (const float*)d_in[9];
  const float* be2 = (const float*)d_in[10];
  float* out = (float*)d_out;

  char* ws = (char*)d_ws;
  _Float16* WT1 = (_Float16*)ws;                     // 512*256*2 = 262144 B
  _Float16* WT2 = (_Float16*)(ws + 262144);          //  64*256*2 =  32768 B
  _Float16* ef  = (_Float16*)(ws + 262144 + 32768);  // 50000*64*2 = 6.4 MB

  prep_weights_kernel<<<576, 256, 0, stream>>>(Wn1, We1, We2, WT1, WT2);
  fused_node_kernel<<<(N_NODES + BM - 1) / BM, 256, 0, stream>>>(
      X, WT1, WT2, bn1, Wn2, bn2, be1, be2, out, ef);
  edge_kernel<<<(N_EDGES * 4 + 255) / 256, 256, 0, stream>>>(ids, ef, out);
}
